// GraphNN_11450382811372
// MI455X (gfx1250) — matrix operand order repaired and checked by reading, not compiled, not run on hardware
//
#include <hip/hip_runtime.h>
#include <hip/hip_bf16.h>

// ---------------------------------------------------------------------------
// GraphNN (GNN w/ learned attention operator) for MI455X / gfx1250.
// All large GEMMs use v_wmma_f32_16x16x32_f16 (f16 in, f32 accumulate).
// B-operands are pre-packed into per-lane fragment layout so each fragment
// is 2x global_load_b128; A-fragments are built from 2x 16B vector loads.
// GEMM inner loops are software-pipelined with ping-pong fragment buffers
// (unrolled x2) so WMMAs issue behind partial s_wait_loadcnt, with no
// rotation copies.
// Pipeline per layer:
//   kPACK (weights -> f16 fragment layout), kCVT (x -> f16)
//   kA   : pairwise |x_n - x_m| GEMM1 -> h1pre (f16, 262144x256)
//   kR+kFS : batch-norm stats -> scale/shift (f32 + f16)
//   kB   : BN+LReLU (packed f16) fused + GEMM2 -> h2pre (f16, 262144x128)
//   kR+kFS : stats
//   kC   : BN+LReLU + dot(wl) -> scores
//   kSM  : masked softmax (wave32 shfl reductions) -> att (f16)
//   kG1  : att @ x (WMMA)        -> y1 (f16)
//   kG2  : [x,y1] @ W + b (WMMA) -> ypre (f32, 2048xFout)
//   kR3  : small BN stats (block reduce)
//   kA3  : BN+LReLU+concat -> next x  (last layer -> d_out)
// ---------------------------------------------------------------------------

typedef __attribute__((ext_vector_type(16))) _Float16       v16h;
typedef __attribute__((ext_vector_type(8)))  _Float16       h8;
typedef __attribute__((ext_vector_type(8)))  float          v8f;
typedef __attribute__((ext_vector_type(8)))  unsigned short us8;

#define EPS   1e-5f
#define SLOPE 0.01f

__device__ __forceinline__ float lrelu(float v) { return v >= 0.f ? v : SLOPE * v; }

__device__ __forceinline__ h8 habs8(h8 x) {
  us8 u = __builtin_bit_cast(us8, x);
  u = u & (unsigned short)0x7fff;
  return __builtin_bit_cast(h8, u);
}

__device__ __forceinline__ v16h cat16(h8 lo, h8 hi) {
  return __builtin_shufflevector(lo, hi, 0, 1, 2, 3, 4, 5, 6, 7,
                                 8, 9, 10, 11, 12, 13, 14, 15);
}

// ---------------- fragment packing --------------------------------------
// B-fragment layout (ISA 7.12.2, 16-bit B 32x16): lane = col + 16*((K>>3)&1),
// element i holds K = 8*(lane>>4) + (i&7) + 16*(i>>3). Packed so one fragment = one aligned
// 32-byte chunk per lane: pk[((kt*CT + ct)*32 + lane)*16 + i]
__global__ void kPACK_b(const float* __restrict__ w, _Float16* __restrict__ pk,
                        int K, int C) {
  const int CT = C >> 4;
  const long total = (long)K * C;
  for (long idx = blockIdx.x * blockDim.x + threadIdx.x; idx < total;
       idx += (long)gridDim.x * blockDim.x) {
    int  i    = (int)(idx & 15);
    int  lane = (int)((idx >> 4) & 31);
    long t    = idx >> 9;
    int  ct   = (int)(t % CT);
    int  kt   = (int)(t / CT);
    int  kk   = kt * 32 + ((lane >> 4) << 3) + (i & 7) + ((i >> 3) << 4);
    int  cc   = ct * 16 + (lane & 15);
    pk[idx] = (_Float16)w[(long)kk * C + cc];
  }
}

// same but f16 source, batched (for x as B-operand of att@x)
__global__ void kPACK_bh(const _Float16* __restrict__ src, _Float16* __restrict__ pk,
                         int K, int C, int nb) {
  const int CT = C >> 4;
  const long per = (long)K * C;
  const long total = per * nb;
  for (long idx = blockIdx.x * blockDim.x + threadIdx.x; idx < total;
       idx += (long)gridDim.x * blockDim.x) {
    long b = idx / per;
    long r = idx - b * per;
    int  i    = (int)(r & 15);
    int  lane = (int)((r >> 4) & 31);
    long t    = r >> 9;
    int  ct   = (int)(t % CT);
    int  kt   = (int)(t / CT);
    int  kk   = kt * 32 + ((lane >> 4) << 3) + (i & 7) + ((i >> 3) << 4);
    int  cc   = ct * 16 + (lane & 15);
    pk[idx] = src[b * per + (long)kk * C + cc];
  }
}

// ---------------- kernel A: pairwise-diff GEMM1 ----------------------------
// grid (4, N, B); block 256 (2x4 waves); 32 pair-rows x 256 channels.
// K-loop unrolled x2 with ping-pong fragment buffers (KT = F/32 is even):
// the even half consumes buffer 0 while preloading buffer 1, and vice versa.
__global__ __launch_bounds__(256)
void kA_pairgemm(const _Float16* __restrict__ xh, const _Float16* __restrict__ w1p,
                 const float* __restrict__ b1, _Float16* __restrict__ h1, int F) {
  const int tid = threadIdx.x, lane = tid & 31, wave = tid >> 5;
  const int wr = wave >> 2, wc = wave & 3;
  const int lane_hi = lane >> 4, lm = lane & 15;
  const int b = blockIdx.z, n = blockIdx.y, m_base = blockIdx.x * 32;

  __shared__ alignas(16) _Float16 lxm[32 * 192];
  __shared__ alignas(16) _Float16 lxn[192];

  const _Float16* xb = xh + (size_t)b * 128 * F;
  {
    const uint4* srcm = (const uint4*)(xb + (size_t)m_base * F);
    uint4* dm = (uint4*)lxm;
    for (int j = tid; j < 4 * F; j += 256) dm[j] = srcm[j];   // 32*F/8 chunks
    const uint4* srcn = (const uint4*)(xb + (size_t)n * F);
    uint4* dn = (uint4*)lxn;
    for (int j = tid; j < F / 8; j += 256) dn[j] = srcn[j];
  }
  __syncthreads();

  const int  mloc = wr * 16 + lm;
  const h8*  lxmv = (const h8*)(lxm + mloc * F);
  const h8*  lxnv = (const h8*)lxn;
  const v16h* w1v = (const v16h*)w1p;            // CT = 16

  v8f acc[4];
#pragma unroll
  for (int t = 0; t < 4; ++t) {
    float bias = b1[wc * 64 + t * 16 + lm];
#pragma unroll
    for (int v = 0; v < 8; ++v) acc[t][v] = bias;
  }

  const int KT = F >> 5;                 // 2, 4 or 6 (always even)
  v16h b0[4], b1f[4];
  h8 exm0, exm1, exn0, exn1, oxm0, oxm1, oxn0, oxn1;
#pragma unroll
  for (int t = 0; t < 4; ++t) b0[t] = w1v[(size_t)((wc * 4 + t) * 32 + lane)];
  exm0 = lxmv[lane_hi]; exm1 = lxmv[2 + lane_hi];
  exn0 = lxnv[lane_hi]; exn1 = lxnv[2 + lane_hi];

  for (int kt = 0; kt < KT; kt += 2) {
    {   // preload odd iteration (kt+1) while even data is consumed
      const int c1 = (kt + 1) * 4 + lane_hi;
      oxm0 = lxmv[c1]; oxm1 = lxmv[c1 + 2];
      oxn0 = lxnv[c1]; oxn1 = lxnv[c1 + 2];
#pragma unroll
      for (int t = 0; t < 4; ++t)
        b1f[t] = w1v[(size_t)(((kt + 1) * 16 + wc * 4 + t) * 32 + lane)];
    }
    v16h a0 = cat16(habs8(exn0 - exm0), habs8(exn1 - exm1));
#pragma unroll
    for (int t = 0; t < 4; ++t)
      acc[t] = __builtin_amdgcn_wmma_f32_16x16x32_f16(false, a0, false, b0[t], (short)0,
                                                      acc[t], false, false);
    if (kt + 2 < KT) {  // preload next even iteration
      const int c2 = (kt + 2) * 4 + lane_hi;
      exm0 = lxmv[c2]; exm1 = lxmv[c2 + 2];
      exn0 = lxnv[c2]; exn1 = lxnv[c2 + 2];
#pragma unroll
      for (int t = 0; t < 4; ++t)
        b0[t] = w1v[(size_t)(((kt + 2) * 16 + wc * 4 + t) * 32 + lane)];
    }
    v16h a1 = cat16(habs8(oxn0 - oxm0), habs8(oxn1 - oxm1));
#pragma unroll
    for (int t = 0; t < 4; ++t)
      acc[t] = __builtin_amdgcn_wmma_f32_16x16x32_f16(false, a1, false, b1f[t], (short)0,
                                                      acc[t], false, false);
  }

  const size_t rbase = (size_t)(b * 128 + n) * 128;
#pragma unroll
  for (int t = 0; t < 4; ++t) {
    const int col = wc * 64 + t * 16 + lm;
#pragma unroll
    for (int v = 0; v < 8; ++v) {
      int m = m_base + wr * 16 + v + (lane_hi ? 8 : 0);
      h1[(rbase + m) * 256 + col] = (_Float16)acc[t][v];
    }
  }
}

// ---------------- kernel B: BN1+LReLU (packed f16) + GEMM2 -----------------
// grid Mpairs/32; block 256 (2x4 waves); 32 rows x 128 cols. Pipelined;
// constant trip count 8 fully unrolled so double-buffer copies rename away.
__global__ __launch_bounds__(256)
void kB_bngemm(const _Float16* __restrict__ h1, const _Float16* __restrict__ w2p,
               const _Float16* __restrict__ scale1h, const _Float16* __restrict__ shift1h,
               const float* __restrict__ b2, _Float16* __restrict__ h2) {
  const int tid = threadIdx.x, lane = tid & 31, wave = tid >> 5;
  const int wr = wave >> 2, wc = wave & 3;
  const int lane_hi = lane >> 4, lm = lane & 15;

  const size_t r = (size_t)blockIdx.x * 32 + wr * 16 + lm;
  const h8* rv  = (const h8*)(h1 + r * 256);
  const h8* scv = (const h8*)scale1h;
  const h8* shv = (const h8*)shift1h;
  const v16h* w2v = (const v16h*)w2p;            // CT = 8

  const _Float16 sl = (_Float16)SLOPE;
  const h8 slope8 = {sl, sl, sl, sl, sl, sl, sl, sl};

  v8f acc[2];
#pragma unroll
  for (int t = 0; t < 2; ++t) {
    float bias = b2[wc * 32 + t * 16 + lm];
#pragma unroll
    for (int v = 0; v < 8; ++v) acc[t][v] = bias;
  }

  // prologue: kt = 0
  v16h bcur[2], bnext[2];
#pragma unroll
  for (int t = 0; t < 2; ++t) bcur[t] = w2v[(size_t)((wc * 2 + t) * 32 + lane)];
  h8 ra0 = rv[lane_hi], ra1 = rv[2 + lane_hi];

#pragma unroll
  for (int kt = 0; kt < 8; ++kt) {
    h8 ra0n, ra1n;
    if (kt < 7) {                      // issue next iteration's loads first
      const int c1 = (kt + 1) * 4 + lane_hi;
      ra0n = rv[c1];
      ra1n = rv[c1 + 2];
#pragma unroll
      for (int t = 0; t < 2; ++t)
        bnext[t] = w2v[(size_t)(((kt + 1) * 8 + wc * 2 + t) * 32 + lane)];
      __builtin_prefetch(h1 + (r + 32) * 256 + kt * 32, 0, 0);
    }
    const int c0 = kt * 4 + lane_hi;
    h8 v0 = ra0 * scv[c0]     + shv[c0];
    h8 v1 = ra1 * scv[c0 + 2] + shv[c0 + 2];
    v0 = __builtin_elementwise_max(v0, v0 * slope8);     // lrelu = max(v, 0.01v)
    v1 = __builtin_elementwise_max(v1, v1 * slope8);
    v16h a = cat16(v0, v1);
#pragma unroll
    for (int t = 0; t < 2; ++t)
      acc[t] = __builtin_amdgcn_wmma_f32_16x16x32_f16(false, a, false, bcur[t], (short)0,
                                                      acc[t], false, false);
    if (kt < 7) {
      ra0 = ra0n; ra1 = ra1n;
#pragma unroll
      for (int t = 0; t < 2; ++t) bcur[t] = bnext[t];
    }
  }

  const size_t rb = (size_t)blockIdx.x * 32 + wr * 16;
#pragma unroll
  for (int t = 0; t < 2; ++t) {
    const int col = wc * 32 + t * 16 + lm;
#pragma unroll
    for (int v = 0; v < 8; ++v)
      h2[(rb + v + (lane_hi ? 8 : 0)) * 128 + col] = (_Float16)acc[t][v];
  }
}

// ---------------- per-channel sum/sumsq over big f16 matrix ----------------
__global__ __launch_bounds__(256)
void kR_stats(const _Float16* __restrict__ d, long rows, int C,
              float* __restrict__ sums, float* __restrict__ sqs) {
  const int c = threadIdx.x % C;
  const int g = threadIdx.x / C;
  const int gs = 256 / C;
  const long per = rows / gridDim.x;
  const long r0 = (long)blockIdx.x * per;
  float s = 0.f, q = 0.f;
  for (long r = r0 + g; r < r0 + per; r += gs) {
    float v = (float)d[(size_t)r * C + c];
    s += v; q += v * v;
  }
  atomicAdd(&sums[c], s);
  atomicAdd(&sqs[c], q);
}

__global__ void kFS(const float* __restrict__ sums, const float* __restrict__ sqs,
                    const float* __restrict__ g, const float* __restrict__ be,
                    float inv_n, int C, float* __restrict__ scale, float* __restrict__ shift,
                    _Float16* __restrict__ scaleh, _Float16* __restrict__ shifth) {
  int c = threadIdx.x;
  if (c < C) {
    float m  = sums[c] * inv_n;
    float v  = sqs[c] * inv_n - m * m;
    float sc = g[c] * rsqrtf(v + EPS);
    float sh = be[c] - m * sc;
    scale[c] = sc;
    shift[c] = sh;
    if (scaleh) { scaleh[c] = (_Float16)sc; shifth[c] = (_Float16)sh; }
  }
}

__global__ void kZ(float* __restrict__ p, int n) {
  int i = blockIdx.x * blockDim.x + threadIdx.x;
  if (i < n) p[i] = 0.f;
}

__global__ void kCVT(const float* __restrict__ in, _Float16* __restrict__ out, long n) {
  for (long i = blockIdx.x * blockDim.x + threadIdx.x; i < n; i += (long)gridDim.x * blockDim.x)
    out[i] = (_Float16)in[i];
}

// ---------------- edge scores: BN2+LReLU + dot(wl) -------------------------
__global__ __launch_bounds__(256)
void kC_score(const _Float16* __restrict__ h2, const float* __restrict__ scale2,
              const float* __restrict__ shift2, const float* __restrict__ wl,
              const float* __restrict__ bl, float* __restrict__ scores) {
  __shared__ float sc[128], sh[128], w[128];
  const int tid = threadIdx.x;
  if (tid < 128) { sc[tid] = scale2[tid]; sh[tid] = shift2[tid]; w[tid] = wl[tid]; }
  __syncthreads();
  const size_t r = (size_t)blockIdx.x * blockDim.x + tid;
  const _Float16* row = h2 + r * 128;
  float acc = bl[0];
#pragma unroll 8
  for (int c = 0; c < 128; ++c) {
    float v = (float)row[c] * sc[c] + sh[c];
    acc += lrelu(v) * w[c];
  }
  scores[r] = acc;
}

// ---------------- masked softmax over m (wave per row) ---------------------
__global__ __launch_bounds__(256)
void kSM_softmax(const float* __restrict__ scores, _Float16* __restrict__ atth) {
  const int wave = threadIdx.x >> 5, lane = threadIdx.x & 31;
  const int row = blockIdx.x * 8 + wave;     // b*128 + n
  const int n = row & 127;
  const float* s = scores + (size_t)row * 128;
  float v[4], mx = -3.4e38f;
#pragma unroll
  for (int j = 0; j < 4; ++j) {
    int m = lane + 32 * j;
    v[j] = s[m] - (m == n ? 1e8f : 0.f);
    mx = fmaxf(mx, v[j]);
  }
  for (int off = 16; off; off >>= 1) mx = fmaxf(mx, __shfl_xor(mx, off, 32));
  float sum = 0.f;
#pragma unroll
  for (int j = 0; j < 4; ++j) { v[j] = __expf(v[j] - mx); sum += v[j]; }
  for (int off = 16; off; off >>= 1) sum += __shfl_xor(sum, off, 32);
  const float inv = 1.f / sum;
#pragma unroll
  for (int j = 0; j < 4; ++j)
    atth[(size_t)row * 128 + lane + 32 * j] = (_Float16)(v[j] * inv);
}

// ---------------- y1 = att @ x (per batch) via WMMA ------------------------
// grid (F/64, 4, B); block 256 (2x4 waves); 32 rows(n) x 64 cols(f)
// K fixed at 128 -> all 4 B-fragments preloaded.
__global__ __launch_bounds__(256)
void kG1_aggr(const _Float16* __restrict__ atth, const _Float16* __restrict__ xbp,
              _Float16* __restrict__ y1h, int F) {
  const int tid = threadIdx.x, lane = tid & 31, wave = tid >> 5;
  const int wr = wave >> 2, wc = wave & 3;
  const int lane_hi = lane >> 4, lm = lane & 15;
  const int b = blockIdx.z;
  const int n0 = blockIdx.y * 32 + wr * 16;
  const int col = blockIdx.x * 64 + wc * 16 + lm;
  const int CT = F >> 4;
  const int ct = blockIdx.x * 4 + wc;
  const h8* av = (const h8*)(atth + (size_t)(b * 128 + n0 + lm) * 128);
  const v16h* xv = (const v16h*)xbp + (size_t)b * 8 * F;   // 128*F/16 v16h per batch

  v16h bb[4];
#pragma unroll
  for (int kt = 0; kt < 4; ++kt)
    bb[kt] = xv[(size_t)((kt * CT + ct) * 32 + lane)];

  v8f acc = {};
#pragma unroll
  for (int kt = 0; kt < 4; ++kt) {
    const int c0 = kt * 4 + lane_hi;
    v16h a = cat16(av[c0], av[c0 + 2]);
    acc = __builtin_amdgcn_wmma_f32_16x16x32_f16(false, a, false, bb[kt], (short)0,
                                                 acc, false, false);
  }
#pragma unroll
  for (int v = 0; v < 8; ++v)
    y1h[(size_t)(b * 128 + n0 + v + (lane_hi ? 8 : 0)) * F + col] = (_Float16)acc[v];
}

// ---------------- ypre = [x, y1] @ Wg + b via WMMA -------------------------
// grid 64; block 256 (2x4 waves); 32 rows x Fout cols. FOUT templated.
template <int FOUT>
__global__ __launch_bounds__(256)
void kG2_linear(const _Float16* __restrict__ xh, const _Float16* __restrict__ y1h,
                const _Float16* __restrict__ wgp, const float* __restrict__ bias,
                float* __restrict__ ypre, int F) {
  constexpr int NT = FOUT / 64;
  constexpr int CT = FOUT / 16;
  const int tid = threadIdx.x, lane = tid & 31, wave = tid >> 5;
  const int wr = wave >> 2, wc = wave & 3;
  const int lane_hi = lane >> 4, lm = lane & 15;
  const int wcb = wc * (FOUT / 4);
  const size_t r = (size_t)blockIdx.x * 32 + wr * 16 + lm;
  const v16h* wv = (const v16h*)wgp;

  v8f acc[NT];
#pragma unroll
  for (int t = 0; t < NT; ++t) {
    float bv = bias[wcb + t * 16 + lm];
#pragma unroll
    for (int v = 0; v < 8; ++v) acc[t][v] = bv;
  }

  const int K = 2 * F;
  for (int f0 = 0; f0 < K; f0 += 32) {
    const _Float16* arow = (f0 < F) ? (xh + r * F + f0) : (y1h + r * F + (f0 - F));
    const h8* av = (const h8*)arow;
    v16h a = cat16(av[lane_hi], av[2 + lane_hi]);
    const int kt = f0 >> 5;
#pragma unroll
    for (int t = 0; t < NT; ++t) {
      v16h bb = wv[(size_t)((kt * CT + wc * NT + t) * 32 + lane)];
      acc[t] = __builtin_amdgcn_wmma_f32_16x16x32_f16(false, a, false, bb, (short)0,
                                                      acc[t], false, false);
    }
  }

  const size_t rb = (size_t)blockIdx.x * 32 + wr * 16;
#pragma unroll
  for (int t = 0; t < NT; ++t) {
    const int col = wcb + t * 16 + lm;
#pragma unroll
    for (int v = 0; v < 8; ++v)
      ypre[(rb + v + (lane_hi ? 8 : 0)) * FOUT + col] = acc[t][v];
  }
}

// ---------------- small BN stats over 2048 rows (block per channel) --------
__global__ __launch_bounds__(256)
void kR3(const float* __restrict__ ypre, const float* __restrict__ g,
         const float* __restrict__ be, int Fout,
         float* __restrict__ scale, float* __restrict__ shift) {
  const int c = blockIdx.x;
  __shared__ float ss[256], qq[256];
  float s = 0.f, q = 0.f;
  for (int r = threadIdx.x; r < 2048; r += 256) {
    float v = ypre[(size_t)r * Fout + c];
    s += v; q += v * v;
  }
  ss[threadIdx.x] = s; qq[threadIdx.x] = q;
  __syncthreads();
  for (int off = 128; off; off >>= 1) {
    if (threadIdx.x < off) {
      ss[threadIdx.x] += ss[threadIdx.x + off];
      qq[threadIdx.x] += qq[threadIdx.x + off];
    }
    __syncthreads();
  }
  if (threadIdx.x == 0) {
    float m = ss[0] * (1.f / 2048.f);
    float v = qq[0] * (1.f / 2048.f) - m * m;
    float sc = g[c] * rsqrtf(v + EPS);
    scale[c] = sc;
    shift[c] = be[c] - m * sc;
  }
}

// ---------------- BN+LReLU + concat -> next x (or final d_out) -------------
__global__ void kA3(const float* __restrict__ xcur, const float* __restrict__ ypre,
                    const float* __restrict__ scale, const float* __restrict__ shift,
                    float* __restrict__ xnext, int F, int Fout) {
  const int FT = F + Fout;
  const long total = 2048L * FT;
  for (long idx = blockIdx.x * blockDim.x + threadIdx.x; idx < total;
       idx += (long)gridDim.x * blockDim.x) {
    long row = idx / FT;
    int  fo  = (int)(idx - row * FT);
    float v;
    if (fo < F) {
      v = xcur[row * F + fo];
    } else {
      int c = fo - F;
      v = ypre[row * Fout + c] * scale[c] + shift[c];
      v = lrelu(v);
    }
    xnext[idx] = v;
  }
}

// ---------------------------------------------------------------------------
extern "C" void kernel_launch(void* const* d_in, const int* in_sizes, int n_in,
                              void* d_out, int out_size, void* d_ws, size_t ws_size,
                              hipStream_t stream) {
  (void)in_sizes; (void)n_in; (void)out_size; (void)ws_size;
  const int B = 16, N = 128;
  const long MP = (long)B * N * N;                 // 262144 pair rows
  const int  Fs[3]    = {64, 128, 192};
  const int  Fouts[3] = {64, 64, 128};
  // d_in flattening (setup_inputs insertion order):
  // 0: x; per w-block: w1,b1,g1,be1,w2,b2,g2,be2,wl,bl; per l-block: w,b,g,be
  const int wbase[3] = {1, 15, 29};
  const int lbase[3] = {11, 25, 39};
  auto fp = [&](int i) { return (const float*)d_in[i]; };

  // ---- workspace layout (~209 MB) ----
  char* ws = (char*)d_ws;
  size_t off = 0;
  auto alloc = [&](size_t bytes) -> char* {
    char* p = ws + off;
    off = (off + bytes + 255) & ~(size_t)255;
    return p;
  };
  _Float16* h1     = (_Float16*)alloc((size_t)MP * 256 * 2);   // 134 MB
  _Float16* h2     = (_Float16*)alloc((size_t)MP * 128 * 2);   //  67 MB
  float*    scores = (float*)   alloc((size_t)MP * 4);
  _Float16* atth   = (_Float16*)alloc((size_t)MP * 2);
  float*    xA     = (float*)   alloc((size_t)2048 * 192 * 4);
  float*    xB     = (float*)   alloc((size_t)2048 * 192 * 4);
  _Float16* xh     = (_Float16*)alloc((size_t)2048 * 192 * 2);
  _Float16* xbp    = (_Float16*)alloc((size_t)2048 * 192 * 2); // x packed as B-frags
  _Float16* y1h    = (_Float16*)alloc((size_t)2048 * 192 * 2);
  float*    ypre   = (float*)   alloc((size_t)2048 * 128 * 4);
  _Float16* w1p    = (_Float16*)alloc((size_t)192 * 256 * 2);
  _Float16* w2p    = (_Float16*)alloc((size_t)256 * 128 * 2);
  _Float16* wgp    = (_Float16*)alloc((size_t)384 * 128 * 2);
  float*    stats  = (float*)   alloc((size_t)2560 * 4);
  _Float16* sclh1  = (_Float16*)alloc((size_t)256 * 2);
  _Float16* shfh1  = (_Float16*)alloc((size_t)256 * 2);
  float* sums1 = stats,        * sqs1 = stats + 256;
  float* scl1  = stats + 512,  * shf1 = stats + 768;
  float* sums2 = stats + 1024, * sqs2 = stats + 1280;
  float* scl2  = stats + 1536, * shf2 = stats + 1792;
  float* scl3  = stats + 2048, * shf3 = stats + 2304;

  for (int l = 0; l < 3; ++l) {
    const int F = Fs[l], Fout = Fouts[l];
    const int wb = wbase[l], lb = lbase[l];
    const float* xcur = (l == 0) ? fp(0) : (l == 1 ? xA : xB);

    // weight packing (f32 -> f16 fragment layout) + x conversion/packing
    kPACK_b<<<64, 256, 0, stream>>>(fp(wb + 0), w1p, F, 256);
    kPACK_b<<<64, 256, 0, stream>>>(fp(wb + 4), w2p, 256, 128);
    kPACK_b<<<64, 256, 0, stream>>>(fp(lb + 0), wgp, 2 * F, Fout);
    kCVT<<<256, 256, 0, stream>>>(xcur, xh, 2048L * F);
    kPACK_bh<<<128, 256, 0, stream>>>(xh, xbp, 128, F, B);
    kZ<<<8, 256, 0, stream>>>(stats, 2048);   // zero sum/sumsq accumulators

    // pairwise GEMM1 -> h1pre
    kA_pairgemm<<<dim3(4, N, B), 256, 0, stream>>>(xh, w1p, fp(wb + 1), h1, F);
    // BN1 stats
    kR_stats<<<256, 256, 0, stream>>>(h1, MP, 256, sums1, sqs1);
    kFS<<<1, 256, 0, stream>>>(sums1, sqs1, fp(wb + 2), fp(wb + 3),
                               1.f / (float)MP, 256, scl1, shf1, sclh1, shfh1);
    // BN1+LReLU fused GEMM2 -> h2pre
    kB_bngemm<<<MP / 32, 256, 0, stream>>>(h1, w2p, sclh1, shfh1, fp(wb + 5), h2);
    // BN2 stats
    kR_stats<<<256, 256, 0, stream>>>(h2, MP, 128, sums2, sqs2);
    kFS<<<1, 256, 0, stream>>>(sums2, sqs2, fp(wb + 6), fp(wb + 7),
                               1.f / (float)MP, 128, scl2, shf2, nullptr, nullptr);
    // edge scores + masked softmax
    kC_score<<<MP / 256, 256, 0, stream>>>(h2, scl2, shf2, fp(wb + 8), fp(wb + 9), scores);
    kSM_softmax<<<256, 256, 0, stream>>>(scores, atth);

    // gconv: y1 = att @ x ; ypre = [x, y1] @ W + b
    kG1_aggr<<<dim3(F / 64, 4, B), 256, 0, stream>>>(atth, xbp, y1h, F);
    if (Fout == 64)
      kG2_linear<64><<<64, 256, 0, stream>>>(xh, y1h, wgp, fp(lb + 1), ypre, F);
    else
      kG2_linear<128><<<64, 256, 0, stream>>>(xh, y1h, wgp, fp(lb + 1), ypre, F);
    // small BN + apply
    kR3<<<Fout, 256, 0, stream>>>(ypre, fp(lb + 2), fp(lb + 3), Fout, scl3, shf3);
    if (l < 2) {
      float* xnext = (l == 0) ? xA : xB;
      kA3<<<512, 256, 0, stream>>>(xcur, ypre, scl3, shf3, xnext, F, Fout);
    } else {
      kA3<<<512, 256, 0, stream>>>(xcur, ypre, scl3, shf3, (float*)d_out, 0, Fout);
    }
  }
}
